// AllGatherInterLLGemm_72962904424821
// MI455X (gfx1250) — matrix operand order repaired and checked by reading, not compiled, not run on hardware
//
#include <hip/hip_runtime.h>

typedef __attribute__((ext_vector_type(16))) __bf16 v16bf;
typedef __attribute__((ext_vector_type(8)))  __bf16 v8bf;
typedef __attribute__((ext_vector_type(4)))  __bf16 v4bf;
typedef __attribute__((ext_vector_type(8)))  float  v8f;
typedef __attribute__((ext_vector_type(4)))  float  v4f;

#define MSG    16384
#define ROWS   512          // 2 * world*bs = 512 fused output rows
#define BM     128          // rows per block
#define BN     128          // cols per block
#define BK     32           // K per WMMA step
#define LDK    40           // padded LDS K-stride (in bf16) to spread banks
#define KTILES (MSG / BK)   // 512

__device__ __forceinline__ v16bf load_frag(const __bf16* base, int o0, int o1) {
    v8bf p = *(const v8bf*)(base + o0);
    v8bf q = *(const v8bf*)(base + o1);
    return __builtin_shufflevector(p, q, 0,1,2,3,4,5,6,7,8,9,10,11,12,13,14,15);
}

__global__ __launch_bounds__(256)
void AllGatherInterLLGemm_72962904424821_kernel(const float* __restrict__ x0,
                                                const float* __restrict__ x1,
                                                const float* __restrict__ W,
                                                const float* __restrict__ bias,
                                                float* __restrict__ out) {
    __shared__ __bf16 Ah[2 * BM * LDK];
    __shared__ __bf16 Al[2 * BM * LDK];
    __shared__ __bf16 Bh[2 * BN * LDK];
    __shared__ __bf16 Bl[2 * BN * LDK];

    const int tid  = threadIdx.x;
    const int lane = tid & 31;
    const int wave = tid >> 5;       // 0..7
    const int wr   = wave & 3;       // row group (4)
    const int wc   = wave >> 2;      // col group (2)
    const int rlow = lane & 15;
    const int rsel = lane >> 4;      // 0 or 1

    const int r0 = blockIdx.x * BM;  // output row block (x fastest -> L2 reuse of W panel)
    const int c0 = blockIdx.y * BN;  // output col block

    const float* Abase = (r0 < 256) ? (x0 + (size_t)r0 * MSG)
                                    : (x1 + (size_t)(r0 - 256) * MSG);
    const float* Bbase = W + (size_t)c0 * MSG;

    v8f acc[2][4];
#pragma unroll
    for (int rt = 0; rt < 2; ++rt)
#pragma unroll
        for (int ct = 0; ct < 4; ++ct)
#pragma unroll
            for (int j = 0; j < 8; ++j) acc[rt][ct][j] = 0.0f;

    v4f ra[4], rb[4];   // global-load staging registers (fp32)

    auto fetch = [&](int kt) {
        const size_t kOff = (size_t)kt * BK;
#pragma unroll
        for (int i = 0; i < 4; ++i) {
            const int idx = i * 256 + tid;
            const int row = idx >> 3;
            const int kv  = idx & 7;
            ra[i] = *(const v4f*)(Abase + (size_t)row * MSG + kOff + kv * 4);
            rb[i] = *(const v4f*)(Bbase + (size_t)row * MSG + kOff + kv * 4);
        }
    };

    auto commit = [&](int buf) {
#pragma unroll
        for (int i = 0; i < 4; ++i) {
            const int idx = i * 256 + tid;
            const int row = idx >> 3;
            const int kv  = idx & 7;
            const int off = buf * BM * LDK + row * LDK + kv * 4;
            v4bf ah, al, bh, bl;
#pragma unroll
            for (int j = 0; j < 4; ++j) {
                float xa = ra[i][j];
                __bf16 ha = (__bf16)xa;
                ah[j] = ha;
                al[j] = (__bf16)(xa - (float)ha);
                float xb = rb[i][j];
                __bf16 hb = (__bf16)xb;
                bh[j] = hb;
                bl[j] = (__bf16)(xb - (float)hb);
            }
            *(v4bf*)&Ah[off] = ah;
            *(v4bf*)&Al[off] = al;
            *(v4bf*)&Bh[off] = bh;
            *(v4bf*)&Bl[off] = bl;
        }
    };

    fetch(0);
    commit(0);

    for (int kt = 0; kt < KTILES; ++kt) {
        const int buf = kt & 1;
        __syncthreads();
        if (kt + 1 < KTILES) fetch(kt + 1);

        const __bf16* sAh = &Ah[buf * BM * LDK];
        const __bf16* sAl = &Al[buf * BM * LDK];
        const __bf16* sBh = &Bh[buf * BN * LDK];
        const __bf16* sBl = &Bl[buf * BN * LDK];

        // A fragments (16x32 bf16): lanes 0-15 hold K[0..8)+K[16..24), lanes 16-31 offset +8
        v16bf ahf[2], alf[2];
#pragma unroll
        for (int rt = 0; rt < 2; ++rt) {
            const int rowb = (wr * 32 + rt * 16 + rlow) * LDK;
            ahf[rt] = load_frag(sAh + rowb, rsel * 8, 16 + rsel * 8);
            alf[rt] = load_frag(sAl + rowb, rsel * 8, 16 + rsel * 8);
        }
        // B fragments (32x16 bf16): lane = column, lanes 0-15 hold K[0..8)+K[16..24), lanes 16-31 offset +8
#pragma unroll
        for (int ct = 0; ct < 4; ++ct) {
            const int colb = (wc * 64 + ct * 16 + rlow) * LDK;
            v16bf bhf = load_frag(sBh + colb, rsel * 8, 16 + rsel * 8);
            v16bf blf = load_frag(sBl + colb, rsel * 8, 16 + rsel * 8);
#pragma unroll
            for (int rt = 0; rt < 2; ++rt) {
                // split-precision fp32 emulation: hi*hi + hi*lo + lo*hi, fp32 accumulate
                acc[rt][ct] = __builtin_amdgcn_wmma_f32_16x16x32_bf16(
                    false, ahf[rt], false, bhf, (short)0, acc[rt][ct], false, false);
                acc[rt][ct] = __builtin_amdgcn_wmma_f32_16x16x32_bf16(
                    false, ahf[rt], false, blf, (short)0, acc[rt][ct], false, false);
                acc[rt][ct] = __builtin_amdgcn_wmma_f32_16x16x32_bf16(
                    false, alf[rt], false, bhf, (short)0, acc[rt][ct], false, false);
            }
        }

        if (kt + 1 < KTILES) commit((kt + 1) & 1);
    }

    // Epilogue: bias add + store. C layout: VGPR j -> row j (lanes 0-15) / j+8 (lanes 16-31)
#pragma unroll
    for (int ct = 0; ct < 4; ++ct) {
        const int col = c0 + wc * 64 + ct * 16 + rlow;
        const float bv = bias[col];
#pragma unroll
        for (int rt = 0; rt < 2; ++rt) {
            const int srow = r0 + wr * 32 + rt * 16 + rsel * 8;
#pragma unroll
            for (int j = 0; j < 8; ++j) {
                out[(size_t)(srow + j) * MSG + col] = acc[rt][ct][j] + bv;
            }
        }
    }
}

extern "C" void kernel_launch(void* const* d_in, const int* in_sizes, int n_in,
                              void* d_out, int out_size, void* d_ws, size_t ws_size,
                              hipStream_t stream) {
    (void)in_sizes; (void)n_in; (void)out_size; (void)d_ws; (void)ws_size;
    const float* x0 = (const float*)d_in[0];
    const float* x1 = (const float*)d_in[1];
    const float* W  = (const float*)d_in[2];
    const float* b  = (const float*)d_in[3];
    float* out = (float*)d_out;

    dim3 grid(ROWS / BM, MSG / BN);  // (4, 128); row-blocks fastest for L2 reuse of W panels
    dim3 block(256);                 // 8 wave32 waves
    AllGatherInterLLGemm_72962904424821_kernel<<<grid, block, 0, stream>>>(x0, x1, W, b, out);
}
